// CustomCausalAttention_65300682768697
// MI455X (gfx1250) — compile-verified, matrix operand order checked by reading, not run on hardware
//
#include <hip/hip_runtime.h>
#include <hip/hip_bf16.h>

typedef __bf16 bf16_t;
typedef __attribute__((ext_vector_type(16))) __bf16 v16bf;
typedef __attribute__((ext_vector_type(8)))  float  v8f;

#define L_TOK   1152
#define D_MODEL 1024
#define D3      3072
#define NPC     1024
#define NHEADS  16
#define DHEAD   64
#define LDP     40   // padded LDS row stride (elements): 80B = 20 banks, conflict-free

// async direct-to-LDS 16B copy (GLOBAL_LOAD_ASYNC_TO_LDS_B128, ASYNCcnt).
// LDS operand = low 32 bits of the generic shared pointer (flat aperture maps
// addr[31:0] to the wave-relative LDS offset, ISA 10.2).
static __device__ __forceinline__ void async_copy_b128(bf16_t* lds_dst, const bf16_t* gsrc) {
    const unsigned loff = (unsigned)(unsigned long long)lds_dst;
    asm volatile("global_load_async_to_lds_b128 %0, %1, off"
                 :: "v"(loff), "v"(gsrc) : "memory");
}
static __device__ __forceinline__ void wait_async0() {
    asm volatile("s_wait_asynccnt 0" ::: "memory");
}

// ---------------------------------------------------------------- convert ----
__global__ void cvt_f32_bf16(const float* __restrict__ in, bf16_t* __restrict__ out, int n) {
    int i = blockIdx.x * blockDim.x + threadIdx.x;
    if (i < n) out[i] = (bf16_t)in[i];
}

// ------------------------------------------------------------- tiled GEMM ----
// C[M][ldc] = A[M][K] @ B[N][K]^T  (A,B bf16 row-major along K; K % 32 == 0)
// block tile 128x64, 4 waves, wave w -> rows w*32..w*32+31 x all 64 columns.
// double-buffered async-to-LDS staging overlapped with WMMA compute.
template <bool OUT_F32>
__global__ __launch_bounds__(128)
void gemm_bt(const bf16_t* __restrict__ A, const bf16_t* __restrict__ B,
             void* __restrict__ C, int K, int ldc) {
    __shared__ __align__(16) bf16_t lsA[2][128 * LDP];
    __shared__ __align__(16) bf16_t lsB[2][64 * LDP];

    const int tid  = threadIdx.x;
    const int wave = tid >> 5;
    const int lane = tid & 31;
    const int half = lane >> 4;
    const int l16  = lane & 15;
    const int m_blk = blockIdx.y * 128;
    const int n_blk = blockIdx.x * 64;

    v8f acc[2][4] = {};

    auto stage = [&](int kb, int buf) {
        // A: 128 rows x 32 cols -> 512 b128 chunks; B: 64 rows -> 256 chunks
#pragma unroll
        for (int c = tid; c < 512; c += 128) {
            const int r = c >> 2, g = c & 3;
            async_copy_b128(&lsA[buf][r * LDP + g * 8],
                            &A[(size_t)(m_blk + r) * K + kb + g * 8]);
        }
#pragma unroll
        for (int c = tid; c < 256; c += 128) {
            const int r = c >> 2, g = c & 3;
            async_copy_b128(&lsB[buf][r * LDP + g * 8],
                            &B[(size_t)(n_blk + r) * K + kb + g * 8]);
        }
    };

    stage(0, 0);
    wait_async0();
    __syncthreads();

    int buf = 0;
    for (int kb = 0; kb < K; kb += 32) {
        if (kb + 32 < K) stage(kb + 32, buf ^ 1);   // prefetch next tile (async)

        // A fragments: lane holds row m, K chunks {half*8..+7} and {16+half*8..+7}
        v16bf af[2];
#pragma unroll
        for (int mi = 0; mi < 2; ++mi) {
            const bf16_t* aRow = &lsA[buf][(wave * 32 + mi * 16 + l16) * LDP];
            ((uint4*)&af[mi])[0] = *(const uint4*)(aRow + half * 8);
            ((uint4*)&af[mi])[1] = *(const uint4*)(aRow + 16 + half * 8);
        }
#pragma unroll
        for (int u = 0; u < 4; ++u) {
            const bf16_t* bRow = &lsB[buf][(u * 16 + l16) * LDP];
            v16bf bf;
            ((uint4*)&bf)[0] = *(const uint4*)(bRow + half * 8);
            ((uint4*)&bf)[1] = *(const uint4*)(bRow + 16 + half * 8);
#pragma unroll
            for (int mi = 0; mi < 2; ++mi)
                acc[mi][u] = __builtin_amdgcn_wmma_f32_16x16x32_bf16(
                    false, af[mi], false, bf, (short)0, acc[mi][u], false, false);
        }

        wait_async0();      // prefetched tile resident in LDS
        __syncthreads();    // all waves done reading buf / writing buf^1
        buf ^= 1;
    }

    // epilogue: C fragment -> lane n = l16, row m = r + 8*half
#pragma unroll
    for (int mi = 0; mi < 2; ++mi)
#pragma unroll
        for (int u = 0; u < 4; ++u)
#pragma unroll
            for (int r = 0; r < 8; ++r) {
                const size_t idx = (size_t)(m_blk + wave * 32 + mi * 16 + r + 8 * half) * ldc
                                 + n_blk + u * 16 + l16;
                if (OUT_F32) ((float*)C)[idx]  = acc[mi][u][r];
                else         ((bf16_t*)C)[idx] = (bf16_t)acc[mi][u][r];
            }
}

// ------------------------------------------------------------- QKV split ----
// qkv[l][3072] -> q[h][l][64], k[h][l][64], vT[h][64][l]
__global__ void qkv_split(const bf16_t* __restrict__ qkv, bf16_t* __restrict__ qb,
                          bf16_t* __restrict__ kb, bf16_t* __restrict__ vtb) {
    int idx = blockIdx.x * blockDim.x + threadIdx.x;
    if (idx >= L_TOK * D3) return;
    const int l = idx / D3;
    const int e = idx - l * D3;
    const bf16_t v = qkv[idx];
    const int which = e >> 10;          // 0=q 1=k 2=v
    const int em = e & 1023;
    const int h  = em >> 6;
    const int dh = em & 63;
    if (which == 0)      qb[(size_t)(h * L_TOK + l) * DHEAD + dh] = v;
    else if (which == 1) kb[(size_t)(h * L_TOK + l) * DHEAD + dh] = v;
    else                 vtb[(size_t)(h * DHEAD + dh) * L_TOK + l] = v;
}

// --------------------------------------------------- flash attention core ----
// one wave per (16-query tile, head). pc queries (i0<1024): keys 0..1023, no
// mask. joint queries: keys 0..1023 plus causal joints, per-element mask only
// in chunks touching j >= 1024.
__global__ __launch_bounds__(32)
void attn_kernel(const bf16_t* __restrict__ qb, const bf16_t* __restrict__ kb,
                 const bf16_t* __restrict__ vtb, bf16_t* __restrict__ ob) {
    __shared__ __align__(16) bf16_t lsP[16 * LDP];

    const int lane = threadIdx.x & 31;
    const int half = lane >> 4;
    const int l16  = lane & 15;
    const int i0 = blockIdx.x * 16;
    const int h  = blockIdx.y;

    const bf16_t* q  = qb  + (size_t)(h * L_TOK + i0) * DHEAD;
    const bf16_t* kk = kb  + (size_t)h * L_TOK * DHEAD;
    const bf16_t* vt = vtb + (size_t)h * DHEAD * L_TOK;

    // Q fragments for the two K=32 d-chunks
    v16bf qf[2];
#pragma unroll
    for (int c = 0; c < 2; ++c) {
        const bf16_t* p = q + l16 * DHEAD + c * 32 + half * 8;
        ((uint4*)&qf[c])[0] = *(const uint4*)(p);
        ((uint4*)&qf[c])[1] = *(const uint4*)(p + 16);
    }

    const float qk_scale = 0.125f * 1.44269504088896340736f;  // 1/sqrt(64) * log2(e)

    float m_run[8], l_run[8];
#pragma unroll
    for (int r = 0; r < 8; ++r) { m_run[r] = -1e30f; l_run[r] = 0.f; }
    v8f o[4] = {};

    const int nchunks = (i0 < NPC) ? (NPC / 32) : ((i0 + 16 + 31) >> 5);

    for (int t = 0; t < nchunks; ++t) {
        const int j0 = t * 32;

        // prefetch next chunk's K rows into cache (global_prefetch_b8)
        if (t + 1 < nchunks)
            __builtin_prefetch(kk + (size_t)(j0 + 32 + l16) * DHEAD, 0, 1);

        // ---- scores: two 16x16 blocks, reduce over d=64 (2 WMMAs each) ----
        v8f s0 = {}, s1 = {};
#pragma unroll
        for (int c = 0; c < 2; ++c) {
            const bf16_t* p0 = kk + (size_t)(j0 + l16) * DHEAD + c * 32 + half * 8;
            v16bf kf0, kf1;
            ((uint4*)&kf0)[0] = *(const uint4*)(p0);
            ((uint4*)&kf0)[1] = *(const uint4*)(p0 + 16);
            const bf16_t* p1 = p0 + 16 * DHEAD;
            ((uint4*)&kf1)[0] = *(const uint4*)(p1);
            ((uint4*)&kf1)[1] = *(const uint4*)(p1 + 16);
            s0 = __builtin_amdgcn_wmma_f32_16x16x32_bf16(false, qf[c], false, kf0, (short)0, s0, false, false);
            s1 = __builtin_amdgcn_wmma_f32_16x16x32_bf16(false, qf[c], false, kf1, (short)0, s1, false, false);
        }

        const bool need_mask = (i0 >= NPC) && (j0 + 31 >= NPC);

        float p0v[8], p1v[8], rmax[8];
#pragma unroll
        for (int r = 0; r < 8; ++r) {
            float a = s0[r] * qk_scale;
            float b = s1[r] * qk_scale;
            if (need_mask) {
                const int i  = i0 + r + 8 * half;
                const int ja = j0 + l16;
                const int jb = ja + 16;
                if (!((ja < NPC) || (ja <= i))) a = -1e30f;
                if (!((jb < NPC) || (jb <= i))) b = -1e30f;
            }
            p0v[r] = a; p1v[r] = b;
            float mx = fmaxf(a, b);
#pragma unroll
            for (int off = 1; off < 16; off <<= 1)
                mx = fmaxf(mx, __shfl_xor(mx, off, 16));
            rmax[r] = mx;
        }

        // ---- online softmax update ----
#pragma unroll
        for (int r = 0; r < 8; ++r) {
            const float mnew  = fmaxf(m_run[r], rmax[r]);
            const float alpha = exp2f(m_run[r] - mnew);
            const float e0 = exp2f(p0v[r] - mnew);
            const float e1 = exp2f(p1v[r] - mnew);
            p0v[r] = e0; p1v[r] = e1;
            float s = e0 + e1;
#pragma unroll
            for (int off = 1; off < 16; off <<= 1)
                s += __shfl_xor(s, off, 16);
            l_run[r] = l_run[r] * alpha + s;
            m_run[r] = mnew;
#pragma unroll
            for (int u = 0; u < 4; ++u) o[u][r] *= alpha;
        }

        // ---- P: C-layout -> A-layout via LDS ----
        __syncthreads();
#pragma unroll
        for (int r = 0; r < 8; ++r) {
            lsP[(r + 8 * half) * LDP + l16]      = (bf16_t)p0v[r];
            lsP[(r + 8 * half) * LDP + 16 + l16] = (bf16_t)p1v[r];
        }
        __syncthreads();
        v16bf pf;
        const bf16_t* pr = &lsP[l16 * LDP];
        ((uint4*)&pf)[0] = *(const uint4*)(pr + half * 8);
        ((uint4*)&pf)[1] = *(const uint4*)(pr + 16 + half * 8);

        // ---- O += P(16x32) @ V(32x16) for 4 dh chunks (Vt rows contiguous) ----
        const bf16_t* vbase = vt + j0 + half * 8;
#pragma unroll
        for (int u = 0; u < 4; ++u) {
            const bf16_t* pv = vbase + (size_t)(u * 16 + l16) * L_TOK;
            v16bf vf;
            ((uint4*)&vf)[0] = *(const uint4*)(pv);
            ((uint4*)&vf)[1] = *(const uint4*)(pv + 16);
            o[u] = __builtin_amdgcn_wmma_f32_16x16x32_bf16(false, pf, false, vf, (short)0, o[u], false, false);
        }
    }

    // ---- normalize + store [l][h*64+dh] bf16 ----
    float inv_l[8];
#pragma unroll
    for (int r = 0; r < 8; ++r) inv_l[r] = 1.0f / l_run[r];
    bf16_t* outp = ob + (size_t)i0 * D_MODEL + h * DHEAD;
#pragma unroll
    for (int u = 0; u < 4; ++u)
#pragma unroll
        for (int r = 0; r < 8; ++r)
            outp[(size_t)(r + 8 * half) * D_MODEL + u * 16 + l16] = (bf16_t)(o[u][r] * inv_l[r]);
}

// ---------------------------------------------------------------- launch ----
extern "C" void kernel_launch(void* const* d_in, const int* in_sizes, int n_in,
                              void* d_out, int out_size, void* d_ws, size_t ws_size,
                              hipStream_t stream) {
    (void)in_sizes; (void)n_in; (void)out_size; (void)ws_size;
    const float* x     = (const float*)d_in[0];
    const float* w_qkv = (const float*)d_in[1];
    const float* w_fc  = (const float*)d_in[2];
    float* out = (float*)d_out;

    char* ws = (char*)d_ws;
    size_t off = 0;
    auto alloc = [&](size_t bytes) -> void* {
        void* p = ws + off;
        off = (off + bytes + 255) & ~(size_t)255;
        return p;
    };
    bf16_t* xb   = (bf16_t*)alloc((size_t)L_TOK * D_MODEL * 2);
    bf16_t* wqb  = (bf16_t*)alloc((size_t)D3 * D_MODEL * 2);
    bf16_t* wfb  = (bf16_t*)alloc((size_t)D_MODEL * D_MODEL * 2);
    bf16_t* qkvb = (bf16_t*)alloc((size_t)L_TOK * D3 * 2);
    bf16_t* qbuf = (bf16_t*)alloc((size_t)NHEADS * L_TOK * DHEAD * 2);
    bf16_t* kbuf = (bf16_t*)alloc((size_t)NHEADS * L_TOK * DHEAD * 2);
    bf16_t* vtbf = (bf16_t*)alloc((size_t)NHEADS * DHEAD * L_TOK * 2);
    bf16_t* attn = (bf16_t*)alloc((size_t)L_TOK * D_MODEL * 2);

    const int n_x  = L_TOK * D_MODEL;
    const int n_wq = D3 * D_MODEL;
    const int n_wf = D_MODEL * D_MODEL;
    cvt_f32_bf16<<<(n_x  + 255) / 256, 256, 0, stream>>>(x,     xb,  n_x);
    cvt_f32_bf16<<<(n_wq + 255) / 256, 256, 0, stream>>>(w_qkv, wqb, n_wq);
    cvt_f32_bf16<<<(n_wf + 255) / 256, 256, 0, stream>>>(w_fc,  wfb, n_wf);

    // QKV projection: [1152,1024] @ [3072,1024]^T -> bf16 [1152,3072]
    gemm_bt<false><<<dim3(D3 / 64, L_TOK / 128), 128, 0, stream>>>(xb, wqb, qkvb, D_MODEL, D3);

    const int n_qkv = L_TOK * D3;
    qkv_split<<<(n_qkv + 255) / 256, 256, 0, stream>>>(qkvb, qbuf, kbuf, vtbf);

    // attention: 72 query tiles x 16 heads, one wave each
    attn_kernel<<<dim3(L_TOK / 16, NHEADS), 32, 0, stream>>>(qbuf, kbuf, vtbf, attn);

    // output projection: [1152,1024] @ [1024,1024]^T -> fp32 d_out
    gemm_bt<true><<<dim3(D_MODEL / 64, L_TOK / 128), 128, 0, stream>>>(attn, wfb, out, D_MODEL, D_MODEL);
}
